// Attention_51737176048525
// MI455X (gfx1250) — compile-verified, matrix operand order checked by reading, not run on hardware
//
#include <hip/hip_runtime.h>

// ---------------------------------------------------------------------------
// Fused multi-head attention for MI455X (gfx1250, wave32, WMMA bf16 + async
// global->LDS copies).   B=4, S=2048, D=1024, H=16, dh=64.
// Stage 1: Q/K/V projections as bf16 WMMA GEMMs, output in [B,H,S,64] bf16.
// Stage 2: flash-attention, double-buffered K tiles via
//          global_load_async_to_lds_b128 (ASYNCcnt), online softmax.
// Workspace: 3 * B*H*S*64 bf16 = 48 MB in d_ws.
// ---------------------------------------------------------------------------

typedef __bf16 bf16_t;
typedef bf16_t bf16x16 __attribute__((ext_vector_type(16)));
typedef float  f32x8   __attribute__((ext_vector_type(8)));

union FragB { bf16x16 v; unsigned u[8]; };

__device__ __forceinline__ f32x8 zero8() {
  f32x8 z = {0.f, 0.f, 0.f, 0.f, 0.f, 0.f, 0.f, 0.f};
  return z;
}

// K-offset (elements) of VGPR r for the 16-bit 16x32 A/B fragment layout:
// k0 = (r>=4 ? 16 : 0) + (lane/16)*8 + (r&3)*2
__device__ __forceinline__ int koff(int r, int hf) {
  return ((r & 4) << 2) + (hf << 3) + ((r & 3) << 1);
}

__device__ __forceinline__ f32x8 wmma_bf16(const FragB& a, const FragB& b, f32x8 c) {
  return __builtin_amdgcn_wmma_f32_16x16x32_bf16(
      /*neg_a=*/false, a.v, /*neg_b=*/false, b.v,
      /*c_mod=*/(short)0, c, /*reuse_a=*/false, /*reuse_b=*/false);
}

// Async 16B copy global -> LDS (tracked by ASYNCcnt).  VDST = LDS byte
// offset (flat LDS aperture low bits), VADDR = 64-bit global address.
__device__ __forceinline__ void async_ld_b128(void* lds_dst, const void* gsrc) {
  unsigned loff = (unsigned)(unsigned long long)(uintptr_t)lds_dst;
  unsigned long long g = (unsigned long long)(uintptr_t)gsrc;
  asm volatile("global_load_async_to_lds_b128 %0, %1, off"
               :: "v"(loff), "v"(g)
               : "memory");
}
__device__ __forceinline__ void wait_async0() {
  asm volatile("s_wait_asynccnt 0x0" ::: "memory");
}
// Same-wave DS ops execute in order (ISA 7.3); this fence stops the compiler
// from reordering LDS accesses around the per-wave P-stripe round trip, and
// the dscnt wait guarantees the stores have landed before the frag loads.
__device__ __forceinline__ void lds_fence_wait() {
  asm volatile("s_wait_dscnt 0x0" ::: "memory");
}
__device__ __forceinline__ void lds_fence() {
  asm volatile("" ::: "memory");
}

// ---------------------------------------------------------------------------
// Stage 1: P = X (8192x1024 fp32) * W (1024x1024 fp32) -> bf16 [B,H,S,64].
// blockIdx.z selects the projection.  128x128 tile / block, 8 waves (4Mx2N).
// ---------------------------------------------------------------------------
__global__ void __launch_bounds__(256)
proj_gemm_kernel(const float* __restrict__ X0, const float* __restrict__ X1,
                 const float* __restrict__ X2,
                 const float* __restrict__ W0, const float* __restrict__ W1,
                 const float* __restrict__ W2,
                 bf16_t* __restrict__ P0, bf16_t* __restrict__ P1,
                 bf16_t* __restrict__ P2) {
  const float* X;
  const float* W;
  bf16_t* P;
  if (blockIdx.z == 0)      { X = X0; W = W0; P = P0; }
  else if (blockIdx.z == 1) { X = X1; W = W1; P = P1; }
  else                      { X = X2; W = W2; P = P2; }

  __shared__ bf16_t As[128][34];  // [m][k], +2 pad breaks bank conflicts
  __shared__ bf16_t Bs[128][34];  // transposed: [n][k]

  const int tid  = threadIdx.x;
  const int wv   = tid >> 5;
  const int lane = tid & 31;
  const int hf   = lane >> 4;
  const int l16  = lane & 15;
  const int mw   = wv >> 1;  // 0..3 -> 32 rows each
  const int nw   = wv & 1;   // 0..1 -> 64 cols each
  const int rowBase = blockIdx.y * 128;
  const int colBase = blockIdx.x * 128;

  f32x8 acc[2][4];
#pragma unroll
  for (int mi = 0; mi < 2; ++mi)
#pragma unroll
    for (int ni = 0; ni < 4; ++ni) acc[mi][ni] = zero8();

  for (int kb = 0; kb < 1024; kb += 32) {
    __syncthreads();
#pragma unroll
    for (int i = 0; i < 16; ++i) {  // A tile 128x32 fp32 -> bf16
      int id = tid + i * 256;
      int r = id >> 5, c = id & 31;
      As[r][c] = (bf16_t)X[(size_t)(rowBase + r) * 1024 + (kb + c)];
    }
#pragma unroll
    for (int i = 0; i < 16; ++i) {  // W tile 32x128, transposed -> Bs[n][k]
      int id = tid + i * 256;
      int kk = id >> 7, n = id & 127;
      Bs[n][kk] = (bf16_t)W[(size_t)(kb + kk) * 1024 + (colBase + n)];
    }
    __syncthreads();

    FragB af[2], bf[4];
#pragma unroll
    for (int mi = 0; mi < 2; ++mi) {
      const int row = mw * 32 + mi * 16 + l16;
#pragma unroll
      for (int r = 0; r < 8; ++r)
        af[mi].u[r] = *(const unsigned*)&As[row][koff(r, hf)];
    }
#pragma unroll
    for (int ni = 0; ni < 4; ++ni) {
      const int col = nw * 64 + ni * 16 + l16;
#pragma unroll
      for (int r = 0; r < 8; ++r)
        bf[ni].u[r] = *(const unsigned*)&Bs[col][koff(r, hf)];
    }
#pragma unroll
    for (int mi = 0; mi < 2; ++mi)
#pragma unroll
      for (int ni = 0; ni < 4; ++ni)
        acc[mi][ni] = wmma_bf16(af[mi], bf[ni], acc[mi][ni]);
  }

  // C layout: lane holds (row = r + 8*half, col = lane%16); write [B,H,S,64].
#pragma unroll
  for (int mi = 0; mi < 2; ++mi)
#pragma unroll
    for (int ni = 0; ni < 4; ++ni)
#pragma unroll
      for (int r = 0; r < 8; ++r) {
        const int gr = rowBase + mw * 32 + mi * 16 + hf * 8 + r;
        const int gc = colBase + nw * 64 + ni * 16 + l16;
        const int bb = gr >> 11, s = gr & 2047;
        const int hh = gc >> 6,  d = gc & 63;
        P[(((size_t)(bb * 16 + hh) * 2048 + s) << 6) + d] = (bf16_t)acc[mi][ni][r];
      }
}

// ---------------------------------------------------------------------------
// Stage 2: flash attention.  One block per (bh, 128-query tile), 8 waves *
// 16 query rows.  K tiles double-buffered through async global->LDS copies;
// V transposed into LDS while the score WMMAs run; per-wave P stripe
// round-trips through LDS with DS-in-order fences (no extra barriers).
// LDS: Ks 2*18.4KB + Vt 16.6KB + Ps 8.7KB = 62.2 KB.
// ---------------------------------------------------------------------------
__global__ void __launch_bounds__(256)
attn_kernel(const bf16_t* __restrict__ qp, const bf16_t* __restrict__ kp,
            const bf16_t* __restrict__ vp, const float* __restrict__ v_mask,
            const float* __restrict__ q_mask, float* __restrict__ out) {
  const int bh    = blockIdx.y;  // b*16 + h
  const int b     = bh >> 4;
  const int h     = bh & 15;
  const int qBase = blockIdx.x * 128;

  const bf16_t* Qg = qp + (size_t)bh * (2048 * 64);
  const bf16_t* Kg = kp + (size_t)bh * (2048 * 64);
  const bf16_t* Vg = vp + (size_t)bh * (2048 * 64);

  // Row stride 72 elems = 144 B = 9*16 B: async b128 writes stay 16-aligned
  // and 36*r mod 64 is a bank permutation over the 16 rows of a B-fragment.
  __shared__ __align__(16) bf16_t Ks[2][128][72];  // [buf][key][dh]
  __shared__ bf16_t Vt[64][130];                   // [dh][key] (transposed)
  __shared__ bf16_t Ps[8][16][34];                 // per-wave P stripe, 32-key chunk

  const int tid  = threadIdx.x;
  const int wv   = tid >> 5;
  const int lane = tid & 31;
  const int hf   = lane >> 4;
  const int l16  = lane & 15;

  // Q A-fragments straight from global (one-time).
  FragB qf[2];
  {
    const size_t row = (size_t)(qBase + wv * 16 + l16) << 6;
#pragma unroll
    for (int ks = 0; ks < 2; ++ks)
#pragma unroll
      for (int r = 0; r < 8; ++r)
        qf[ks].u[r] = *(const unsigned*)&Qg[row + ks * 32 + koff(r, hf)];
  }

  float mrow[8], lrow[8];
#pragma unroll
  for (int r = 0; r < 8; ++r) { mrow[r] = -3.0e38f; lrow[r] = 0.0f; }
  f32x8 o[4];
#pragma unroll
  for (int nt = 0; nt < 4; ++nt) o[nt] = zero8();

  // Prologue: async-copy K tile 0 into buffer 0 (4 x 16B per thread).
#pragma unroll
  for (int i = 0; i < 4; ++i) {
    int id = tid + i * 256;
    int r = id >> 3, c8 = (id & 7) << 3;
    async_ld_b128(&Ks[0][r][c8], Kg + ((size_t)r << 6) + c8);
  }

  for (int jt = 0; jt < 16; ++jt) {
    const int cur = jt & 1;
    const int kb  = jt << 7;

    wait_async0();     // this wave's K copies for buf[cur] have landed
    __syncthreads();   // all waves' copies landed; prev iter fully consumed

    if (jt + 1 < 16) {  // prefetch next K tile into the other buffer
      const int nkb = kb + 128;
#pragma unroll
      for (int i = 0; i < 4; ++i) {
        int id = tid + i * 256;
        int r = id >> 3, c8 = (id & 7) << 3;
        async_ld_b128(&Ks[cur ^ 1][r][c8], Kg + ((size_t)(nkb + r) << 6) + c8);
      }
    }

    // Fill Vt (transposed) for this tile; visibility barrier is below, so the
    // latency hides behind the score WMMAs.
#pragma unroll
    for (int i = 0; i < 8; ++i) {
      int id = tid + i * 256;
      int r = id >> 4, c4 = (id & 15) << 2;
      unsigned long long vv =
          *(const unsigned long long*)&Vg[((size_t)(kb + r) << 6) + c4];
#pragma unroll
      for (int j = 0; j < 4; ++j)
        *(unsigned short*)&Vt[c4 + j][r] = (unsigned short)(vv >> (16 * j));
    }

    // --- scores: S = Q * K^T (16 queries x 128 keys per wave) ---
    f32x8 sc[8];
#pragma unroll
    for (int t = 0; t < 8; ++t) {
      sc[t] = zero8();
      const int key = t * 16 + l16;
      FragB kf;
#pragma unroll
      for (int ks = 0; ks < 2; ++ks) {
#pragma unroll
        for (int r = 0; r < 8; ++r)
          kf.u[r] = *(const unsigned*)&Ks[cur][key][ks * 32 + koff(r, hf)];
        sc[t] = wmma_bf16(qf[ks], kf, sc[t]);
      }
      const float ma = (1.0f - v_mask[(size_t)b * 2048 + kb + key]) * 1.0e10f;
#pragma unroll
      for (int r = 0; r < 8; ++r) sc[t][r] = sc[t][r] * 0.125f - ma;
    }

    // --- online softmax (row = reg + 8*half; cols in the 16-lane half) ---
    float alpha[8];
#pragma unroll
    for (int r = 0; r < 8; ++r) {
      float nm = -3.0e38f;
#pragma unroll
      for (int t = 0; t < 8; ++t) nm = fmaxf(nm, sc[t][r]);
      nm = fmaxf(nm, __shfl_xor(nm, 1));
      nm = fmaxf(nm, __shfl_xor(nm, 2));
      nm = fmaxf(nm, __shfl_xor(nm, 4));
      nm = fmaxf(nm, __shfl_xor(nm, 8));
      const float mn = fmaxf(mrow[r], nm);
      alpha[r] = __expf(mrow[r] - mn);
      mrow[r] = mn;
      float rs = 0.0f;
#pragma unroll
      for (int t = 0; t < 8; ++t) {
        float p = __expf(sc[t][r] - mn);
        sc[t][r] = p;
        rs += p;
      }
      rs += __shfl_xor(rs, 1);
      rs += __shfl_xor(rs, 2);
      rs += __shfl_xor(rs, 4);
      rs += __shfl_xor(rs, 8);
      lrow[r] = lrow[r] * alpha[r] + rs;
    }
#pragma unroll
    for (int nt = 0; nt < 4; ++nt)
#pragma unroll
      for (int r = 0; r < 8; ++r) o[nt][r] *= alpha[r];

    __syncthreads();  // Vt filled by all waves

    // --- O += P * V in four 32-key chunks through the per-wave P stripe.
    // Same-wave DS in-order + fences replace barriers (stripe is private).
#pragma unroll
    for (int ch = 0; ch < 4; ++ch) {
      lds_fence();  // keep stores after previous chunk's loads
#pragma unroll
      for (int t = 0; t < 2; ++t)
#pragma unroll
        for (int r = 0; r < 8; ++r)
          Ps[wv][hf * 8 + r][t * 16 + l16] = (bf16_t)sc[ch * 2 + t][r];
      lds_fence_wait();  // stores landed before fragment loads

      FragB pf;
#pragma unroll
      for (int r = 0; r < 8; ++r)
        pf.u[r] = *(const unsigned*)&Ps[wv][l16][koff(r, hf)];
      const int kb2 = ch << 5;
#pragma unroll
      for (int nt = 0; nt < 4; ++nt) {
        FragB vf;
        const int col = nt * 16 + l16;
#pragma unroll
        for (int r = 0; r < 8; ++r)
          vf.u[r] = *(const unsigned*)&Vt[col][kb2 + koff(r, hf)];
        o[nt] = wmma_bf16(pf, vf, o[nt]);
      }
    }
  }

  // --- epilogue: normalize, apply q_mask, write fp32 [B,S,H*64] ---
#pragma unroll
  for (int r = 0; r < 8; ++r) {
    const int q = qBase + wv * 16 + hf * 8 + r;
    const float qm = q_mask[(size_t)b * 2048 + q];
    const float inv = qm / lrow[r];
#pragma unroll
    for (int nt = 0; nt < 4; ++nt) {
      const int d = nt * 16 + l16;
      out[(((size_t)b * 2048 + q) << 10) + (h << 6) + d] = o[nt][r] * inv;
    }
  }
}

// ---------------------------------------------------------------------------
extern "C" void kernel_launch(void* const* d_in, const int* in_sizes, int n_in,
                              void* d_out, int out_size, void* d_ws,
                              size_t ws_size, hipStream_t stream) {
  const float* q_value = (const float*)d_in[0];
  const float* k_value = (const float*)d_in[1];
  const float* v_value = (const float*)d_in[2];
  const float* v_mask  = (const float*)d_in[3];
  const float* q_mask  = (const float*)d_in[4];
  const float* Wq      = (const float*)d_in[5];
  const float* Wk      = (const float*)d_in[6];
  const float* Wv      = (const float*)d_in[7];
  float* out = (float*)d_out;

  // Workspace: 3 bf16 projection tensors of B*H*S*64 = 8,388,608 elems each.
  const size_t PROJ_ELEMS = (size_t)4 * 16 * 2048 * 64;
  bf16_t* qp = (bf16_t*)d_ws;
  bf16_t* kp = qp + PROJ_ELEMS;
  bf16_t* vp = kp + PROJ_ELEMS;

  dim3 gridP(1024 / 128, 8192 / 128, 3);  // (N tiles, M tiles, {Q,K,V})
  proj_gemm_kernel<<<gridP, 256, 0, stream>>>(q_value, k_value, v_value,
                                              Wq, Wk, Wv, qp, kp, vp);

  dim3 gridA(2048 / 128, 4 * 16);         // (query tiles, b*H)
  attn_kernel<<<gridA, 256, 0, stream>>>(qp, kp, vp, v_mask, q_mask, out);
}
